// CausalSelfAttention_28252294873375
// MI455X (gfx1250) — matrix operand order repaired and checked by reading, not compiled, not run on hardware
//
#include <hip/hip_runtime.h>

#define B_  4
#define T_  2048
#define D_  1024
#define H_  16
#define DK_ 64

typedef __attribute__((ext_vector_type(16))) __bf16 bf16x16;
typedef __attribute__((ext_vector_type(8)))  float  floatx8;

#ifndef __has_builtin
#define __has_builtin(x) 0
#endif
#if __has_builtin(__builtin_amdgcn_tensor_load_to_lds)
#define HAVE_TDM 1
#else
#define HAVE_TDM 0
#endif

union FragU { float4 v[2]; bf16x16 f; };

__device__ inline bf16x16 make_frag(float4 lo, float4 hi) {
  FragU u; u.v[0] = lo; u.v[1] = hi; return u.f;
}
__device__ inline float4 ldg4(const __bf16* p) { return *reinterpret_cast<const float4*>(p); }

__device__ inline floatx8 wmma_bf16(bf16x16 a, bf16x16 b, floatx8 c) {
  // (neg_a, A, neg_b, B, c_mod, C, reuse_a, reuse_b)
  return __builtin_amdgcn_wmma_f32_16x16x32_bf16(false, a, false, b, (short)0, c, false, false);
}

// ---------------------------------------------------------------------------
// TDM: DMA a 64x32 bf16 tile (rows strided by rowStride elems) into LDS with
// hardware padding: pad_interval=3 (every 16 dwords = one 32-elem row),
// pad_amount=3 (4 dwords = 8 elems) -> padded LDS row stride of 40 bf16.
// ---------------------------------------------------------------------------
#if HAVE_TDM
typedef __attribute__((ext_vector_type(4))) unsigned int uint32x4;
typedef __attribute__((ext_vector_type(8))) int int32x8;
typedef __attribute__((ext_vector_type(4))) int int32x4;

__device__ inline void tdm_load_64x32(const __bf16* gsrc, int rowStride, unsigned ldsByte) {
  const unsigned long long ga = (unsigned long long)gsrc;
  // D# group 0: count=1 | lds_addr | global_addr[56:0] | type=2
  uint32x4 g0 = { 1u,
                  ldsByte,
                  (unsigned)(ga & 0xffffffffu),
                  (unsigned)((ga >> 32) & 0x1ffffffu) | (2u << 30) };
  // D# group 1: data_size=1(2B), pad_enable, pad_interval=3(16dw), pad_amount=3(4dw)
  int32x8 g1 = { (int)((1u << 16) | (1u << 20) | (3u << 22) | (3u << 25)),
                 (int)(((unsigned)rowStride & 0xffffu) << 16),     // tensor_dim0 lo16
                 (int)((((unsigned)rowStride >> 16) & 0xffffu)     // tensor_dim0 hi16
                       | (4096u << 16)),                           // tensor_dim1 lo16
                 (int)(32u << 16),                                 // tile_dim0 = 32
                 64,                                               // tile_dim1 = 64
                 rowStride,                                        // dim0_stride lo32
                 0, 0 };
  int32x4 z4 = { 0, 0, 0, 0 };
#if __clang_major__ >= 23
  int32x8 z8 = { 0, 0, 0, 0, 0, 0, 0, 0 };
  __builtin_amdgcn_tensor_load_to_lds(g0, g1, z4, z4, z8, 0);
#else
  __builtin_amdgcn_tensor_load_to_lds(g0, g1, z4, z4, 0);
#endif
}
#endif

// ---------------------------------------------------------------------------
// f32 -> bf16 conversion (vectorized x4)
// ---------------------------------------------------------------------------
__global__ __launch_bounds__(256) void cvt_kernel(const float* __restrict__ src,
                                                  __bf16* __restrict__ dst, int n4) {
  int i = blockIdx.x * 256 + threadIdx.x;
  if (i < n4) {
    float4 v = reinterpret_cast<const float4*>(src)[i];
    union { __bf16 h[4]; uint2 u; } pk;
    pk.h[0] = (__bf16)v.x; pk.h[1] = (__bf16)v.y;
    pk.h[2] = (__bf16)v.z; pk.h[3] = (__bf16)v.w;
    reinterpret_cast<uint2*>(dst)[i] = pk.u;
  }
}

// ---------------------------------------------------------------------------
// 128x64 GEMM mainloop: block = 4 waves, wave w owns rows [mwave, mwave+32).
// B tile (64 rows x 32 k) staged into double-buffered LDS by the TDM
// (wave 0 issues, prefetch-by-one, TENSORcnt pipelined); padded rows of 40.
// ---------------------------------------------------------------------------
__device__ inline void gemm_mainloop_128x64(const __bf16* __restrict__ Amat,
                                            const __bf16* __restrict__ Bmat,
                                            int Kd, int mwave, int nbase,
                                            __bf16* WT, floatx8 acc[2][4]) {
  const int tid  = threadIdx.x;
  const int lane = tid & 31;
  const int ln = lane & 15, hs = lane >> 4;
  const __bf16* arow0 = Amat + (size_t)(mwave + ln) * Kd;
  const __bf16* arow1 = arow0 + (size_t)16 * Kd;
  const __bf16* btile = Bmat + (size_t)nbase * Kd;
  const int niter = Kd / 32;

#if HAVE_TDM
  if (tid < 32) tdm_load_64x32(btile, Kd, (unsigned)(unsigned long long)WT);
#endif
  for (int i = 0; i < niter; ++i) {
    const int k0  = i * 32;
    const int buf = i & 1;
#if HAVE_TDM
    if (tid < 32) {
      if (i + 1 < niter) {
        tdm_load_64x32(btile + k0 + 32, Kd,
                       (unsigned)(unsigned long long)(WT + ((i + 1) & 1) * 2560));
        __builtin_amdgcn_s_wait_tensorcnt(1);   // tile i complete, i+1 in flight
      } else {
        __builtin_amdgcn_s_wait_tensorcnt(0);
      }
    }
#else
    // fallback: cooperative vector staging
#pragma unroll
    for (int rep = 0; rep < 2; ++rep) {
      int c   = tid + rep * 128;
      int row = c >> 2;
      int col = (c & 3) * 8;
      float4 v = ldg4(btile + (size_t)row * Kd + k0 + col);
      *reinterpret_cast<float4*>(WT + buf * 2560 + row * 40 + col) = v;
    }
#endif
    __syncthreads();   // publish buf i to all waves
    bf16x16 a0 = make_frag(ldg4(arow0 + k0 + hs * 8), ldg4(arow0 + k0 + 16 + hs * 8));
    bf16x16 a1 = make_frag(ldg4(arow1 + k0 + hs * 8), ldg4(arow1 + k0 + 16 + hs * 8));
#pragma unroll
    for (int nt = 0; nt < 4; ++nt) {
      const __bf16* brow = WT + buf * 2560 + (nt * 16 + ln) * 40 + hs * 8;
      bf16x16 bfr = make_frag(*reinterpret_cast<const float4*>(brow),
                              *reinterpret_cast<const float4*>(brow + 16));
      acc[0][nt] = wmma_bf16(a0, bfr, acc[0][nt]);
      acc[1][nt] = wmma_bf16(a1, bfr, acc[1][nt]);
    }
    __syncthreads();   // all waves done reading buf i before it is overwritten
  }
}

// ---------------------------------------------------------------------------
// Kernel 1: QKV = x @ Wqkv^T, scatter into Q[b,h,t,d], K[b,h,t,d], Vt[b,h,d,t]
// ---------------------------------------------------------------------------
__global__ __launch_bounds__(128) void qkv_kernel(const __bf16* __restrict__ xbf,
                                                  const __bf16* __restrict__ wbf,
                                                  __bf16* __restrict__ Q,
                                                  __bf16* __restrict__ Kc,
                                                  __bf16* __restrict__ Vt) {
  __shared__ __bf16 WT[2 * 64 * 40];
  const int tid = threadIdx.x, wave = tid >> 5, lane = tid & 31;
  const int ln = lane & 15, hs = lane >> 4;
  const int mwave = blockIdx.x * 128 + wave * 32;
  const int nbase = blockIdx.y * 64;
  floatx8 acc[2][4] = {};
  gemm_mainloop_128x64(xbf, wbf, D_, mwave, nbase, WT, acc);
#pragma unroll
  for (int nt = 0; nt < 4; ++nt) {
    const int colg  = nbase + nt * 16 + ln;
    const int which = colg >> 10;           // 0=q 1=k 2=v
    const int h     = (colg >> 6) & (H_ - 1);
    const int d     = colg & (DK_ - 1);
#pragma unroll
    for (int g = 0; g < 2; ++g)
#pragma unroll
      for (int r = 0; r < 8; ++r) {
        const int mg = mwave + g * 16 + hs * 8 + r;
        const int b  = mg >> 11;            // T_ = 2048
        const int t  = mg & (T_ - 1);
        const __bf16 v = (__bf16)acc[g][nt][r];
        const size_t bh = (size_t)(b * H_ + h);
        if (which == 0)      Q [(bh * T_ + t) * DK_ + d] = v;
        else if (which == 1) Kc[(bh * T_ + t) * DK_ + d] = v;
        else                 Vt[(bh * DK_ + d) * T_ + t] = v;
      }
  }
}

// ---------------------------------------------------------------------------
// Kernel 2: flash attention with prefix-LM mask. 1 wave = one 16-query tile.
// allowed(q,k): k <= (q < prefix ? prefix-1 : q)
// ---------------------------------------------------------------------------
__global__ __launch_bounds__(128) void attn_kernel(const __bf16* __restrict__ Q,
                                                   const __bf16* __restrict__ Km,
                                                   const __bf16* __restrict__ Vt,
                                                   const int* __restrict__ plen,
                                                   __bf16* __restrict__ AO) {
  __shared__ __bf16 PT[4][16 * 72];         // per-wave P staging (padded rows)
  const int tid = threadIdx.x, wave = tid >> 5, lane = tid & 31;
  const int ln = lane & 15, hs = lane >> 4;
  const int idx   = blockIdx.x * 4 + wave;
  const int qt    = idx & (T_ / 16 - 1);
  const int bh    = idx >> 7;               // T_/16 = 128 tiles
  const int qbase = qt * 16;
  const int b     = bh >> 4;
  int pv = plen[b];
  pv = pv < 0 ? 0 : (pv > T_ ? T_ : pv);

  int lim[8];
#pragma unroll
  for (int r = 0; r < 8; ++r) {
    const int q = qbase + hs * 8 + r;
    lim[r] = (q < pv) ? (pv - 1) : q;
  }
  const int maxlim = (qbase + 15 >= pv) ? (qbase + 15) : (pv - 1);

  // Q A-fragments (16x32 x 2 chunks over DK=64)
  const __bf16* qrow = Q + ((size_t)bh * T_ + qbase + ln) * DK_;
  bf16x16 qa[2];
#pragma unroll
  for (int dc = 0; dc < 2; ++dc)
    qa[dc] = make_frag(ldg4(qrow + dc * 32 + hs * 8),
                       ldg4(qrow + dc * 32 + 16 + hs * 8));

  const __bf16* Kb = Km + (size_t)bh * T_ * DK_;
  const __bf16* Vb = Vt + (size_t)bh * DK_ * T_;

  floatx8 o[4] = {};
  float rmax[8], rsum[8];
#pragma unroll
  for (int r = 0; r < 8; ++r) { rmax[r] = -1e30f; rsum[r] = 0.f; }

  __bf16* pp = PT[wave];
  for (int kb = 0; kb <= maxlim; kb += 64) {
    float s[4][8];
    // S = (Q K^T) * scale, masked
#pragma unroll
    for (int jt = 0; jt < 4; ++jt) {
      floatx8 sc = {};
      const __bf16* krow = Kb + (size_t)(kb + jt * 16 + ln) * DK_;
#pragma unroll
      for (int dc = 0; dc < 2; ++dc) {
        bf16x16 kf = make_frag(ldg4(krow + dc * 32 + hs * 8),
                               ldg4(krow + dc * 32 + hs * 8 + 16));
        sc = wmma_bf16(qa[dc], kf, sc);
      }
      const int kk = kb + jt * 16 + ln;
#pragma unroll
      for (int r = 0; r < 8; ++r)
        s[jt][r] = (kk <= lim[r]) ? sc[r] * 0.125f : -1e30f;
    }
    // online softmax per row (rows live on 16-lane halves)
#pragma unroll
    for (int r = 0; r < 8; ++r) {
      float tm = fmaxf(fmaxf(s[0][r], s[1][r]), fmaxf(s[2][r], s[3][r]));
#pragma unroll
      for (int off = 8; off >= 1; off >>= 1)
        tm = fmaxf(tm, __shfl_xor(tm, off, 16));
      const float nm   = fmaxf(rmax[r], tm);
      const float corr = __expf(rmax[r] - nm);
      rmax[r] = nm;
      float ps = 0.f;
#pragma unroll
      for (int jt = 0; jt < 4; ++jt) {
        const float p = __expf(s[jt][r] - nm);
        s[jt][r] = p; ps += p;
      }
#pragma unroll
      for (int off = 8; off >= 1; off >>= 1)
        ps += __shfl_xor(ps, off, 16);
      rsum[r] = rsum[r] * corr + ps;
#pragma unroll
      for (int nt = 0; nt < 4; ++nt) o[nt][r] *= corr;
    }
    // C-layout P -> LDS -> A-fragment layout
#pragma unroll
    for (int jt = 0; jt < 4; ++jt)
#pragma unroll
      for (int r = 0; r < 8; ++r)
        pp[(hs * 8 + r) * 72 + jt * 16 + ln] = (__bf16)s[jt][r];
    // O += P V  (2 K-chunks of 32 keys, 4 N-tiles over DK)
#pragma unroll
    for (int kc = 0; kc < 2; ++kc) {
      const __bf16* prow = pp + ln * 72 + kc * 32 + hs * 8;
      bf16x16 pa = make_frag(*reinterpret_cast<const float4*>(prow),
                             *reinterpret_cast<const float4*>(prow + 16));
#pragma unroll
      for (int nt = 0; nt < 4; ++nt) {
        const __bf16* vrow = Vb + (size_t)(nt * 16 + ln) * T_ + kb + kc * 32 + hs * 8;
        bf16x16 vf = make_frag(ldg4(vrow), ldg4(vrow + 16));
        o[nt] = wmma_bf16(pa, vf, o[nt]);
      }
    }
  }
  // normalize + store to [b, t, h*64 + d] (bf16)
  const int h = bh & (H_ - 1);
#pragma unroll
  for (int r = 0; r < 8; ++r) {
    const float inv = 1.0f / rsum[r];
    const int t = qbase + hs * 8 + r;
    __bf16* orow = AO + ((size_t)b * T_ + t) * D_ + h * DK_;
#pragma unroll
    for (int nt = 0; nt < 4; ++nt)
      orow[nt * 16 + ln] = (__bf16)(o[nt][r] * inv);
  }
}

// ---------------------------------------------------------------------------
// Kernel 3: out = attn @ Wo^T  (fp32 output)
// ---------------------------------------------------------------------------
__global__ __launch_bounds__(128) void oproj_kernel(const __bf16* __restrict__ abf,
                                                    const __bf16* __restrict__ wbf,
                                                    float* __restrict__ out) {
  __shared__ __bf16 WT[2 * 64 * 40];
  const int tid = threadIdx.x, wave = tid >> 5, lane = tid & 31;
  const int ln = lane & 15, hs = lane >> 4;
  const int mwave = blockIdx.x * 128 + wave * 32;
  const int nbase = blockIdx.y * 64;
  floatx8 acc[2][4] = {};
  gemm_mainloop_128x64(abf, wbf, D_, mwave, nbase, WT, acc);
#pragma unroll
  for (int g = 0; g < 2; ++g)
#pragma unroll
    for (int nt = 0; nt < 4; ++nt)
#pragma unroll
      for (int r = 0; r < 8; ++r) {
        const int mg = mwave + g * 16 + hs * 8 + r;
        out[(size_t)mg * D_ + nbase + nt * 16 + ln] = acc[g][nt][r];
      }
}

// ---------------------------------------------------------------------------
extern "C" void kernel_launch(void* const* d_in, const int* in_sizes, int n_in,
                              void* d_out, int out_size, void* d_ws, size_t ws_size,
                              hipStream_t stream) {
  const float* x    = (const float*)d_in[0];
  const int*   pl   = (const int*)d_in[1];   // prefix_lengths
  const float* wqkv = (const float*)d_in[2];
  const float* wo   = (const float*)d_in[3];
  float* out = (float*)d_out;

  char* ws = (char*)d_ws;
  size_t off = 0;
  auto carve = [&](size_t bytes) -> char* {
    char* p = ws + off;
    off += (bytes + 255) & ~(size_t)255;
    return p;
  };
  __bf16* xbf  = (__bf16*)carve((size_t)B_ * T_ * D_ * 2);
  __bf16* wqbf = (__bf16*)carve((size_t)3 * D_ * D_ * 2);
  __bf16* wobf = (__bf16*)carve((size_t)D_ * D_ * 2);
  __bf16* Qbf  = (__bf16*)carve((size_t)B_ * H_ * T_ * DK_ * 2);
  __bf16* Kbf  = (__bf16*)carve((size_t)B_ * H_ * T_ * DK_ * 2);
  __bf16* Vtbf = (__bf16*)carve((size_t)B_ * H_ * DK_ * T_ * 2);
  __bf16* Abf  = (__bf16*)carve((size_t)B_ * T_ * D_ * 2);
  (void)in_sizes; (void)n_in; (void)out_size; (void)ws_size;

  const int nx4 = B_ * T_ * D_ / 4;
  const int nq4 = 3 * D_ * D_ / 4;
  const int no4 = D_ * D_ / 4;
  cvt_kernel<<<(nx4 + 255) / 256, 256, 0, stream>>>(x, xbf, nx4);
  cvt_kernel<<<(nq4 + 255) / 256, 256, 0, stream>>>(wqkv, wqbf, nq4);
  cvt_kernel<<<(no4 + 255) / 256, 256, 0, stream>>>(wo, wobf, no4);

  qkv_kernel<<<dim3(B_ * T_ / 128, 3 * D_ / 64), 128, 0, stream>>>(xbf, wqbf, Qbf, Kbf, Vtbf);
  attn_kernel<<<B_ * H_ * (T_ / 16) / 4, 128, 0, stream>>>(Qbf, Kbf, Vtbf, pl, Abf);
  oproj_kernel<<<dim3(B_ * T_ / 128, D_ / 64), 128, 0, stream>>>(Abf, wobf, out);
}
